// AttentionTeacher_83107617178485
// MI455X (gfx1250) — matrix operand order repaired and checked by reading, not compiled, not run on hardware
//
#include <hip/hip_runtime.h>
#include <hip/hip_bf16.h>

// ---------------------------------------------------------------------------
// AttentionTeacher on MI455X (gfx1250).
//   out = (sum_h softmax_h((x Wq^T + bq)(x Wk^T + bk)^T / 8)) @ V / 16
// All three contractions run on v_wmma_f32_16x16x32_bf16 (f32 accumulate);
// softmax is exact f32. Inputs are pre-converted to bf16 once, GEMM K-loops
// are double-buffered (loads of step k+1 overlap WMMAs of step k).
// ---------------------------------------------------------------------------

typedef __bf16 bf16_t;
typedef __attribute__((ext_vector_type(16))) __bf16 v16bf;
typedef __attribute__((ext_vector_type(8)))  __bf16 v8bf;
typedef __attribute__((ext_vector_type(8)))  float  v8f;

#define B_   4
#define S_   1024
#define HID_ 1024
#define NH_  16
#define HD_  64

// D = A*B + C, 16x16 tile, K=32, bf16 inputs, f32 accumulate
__device__ inline v8f wmma_bf16(v16bf a, v16bf b, v8f c) {
  return __builtin_amdgcn_wmma_f32_16x16x32_bf16(
      /*neg_a=*/false, a, /*neg_b=*/false, b,
      /*c_mod=*/(short)0, c, /*reuse_a=*/false, /*reuse_b=*/false);
}

// A-matrix fragment (16x32 bf16). Lanes 0-15 hold M=0..15 with
// K = 0..15; lanes 16-31 hold same M with K = 16..31.
__device__ inline v16bf a_frag_bf16(const bf16_t* __restrict__ p, int ld,
                                    int m_base, int k_base, int lane) {
  const int m  = m_base + (lane & 15);
  const int o8 = (lane >> 4) << 4;
  const bf16_t* r = p + (long)m * ld + k_base;
  v16bf a;
#pragma unroll
  for (int e = 0; e < 16; ++e) a[e] = r[o8 + e];
  return a;
}

// B-matrix fragment (32x16 bf16) where the contraction index runs
// contiguously along row `n` of the source (source is N x K row-major:
// W, k_proj, vT all have this shape). Lane n holds output column n;
// lanes 0-15 K=0..15, lanes 16-31 K=16..31 -> one 32B load per lane.
__device__ inline v16bf b_frag_rowk_bf16(const bf16_t* __restrict__ p, int ld,
                                         int row_base, int k_base, int lane) {
  const int row = row_base + (lane & 15);
  const int k0  = k_base + ((lane >> 4) << 4);
  const bf16_t* r = p + (long)row * ld + k0;
  v16bf b;
#pragma unroll
  for (int e = 0; e < 16; ++e) b[e] = r[e];
  return b;
}

// ---------------------------------------------------------------------------
// f32 -> bf16 streaming convert, 8 elements/thread (b128 in, b128 out)
// ---------------------------------------------------------------------------
__global__ void k_cvt(const float* __restrict__ src, bf16_t* __restrict__ dst) {
  const long i = ((long)blockIdx.x * blockDim.x + threadIdx.x) * 8;
  v8bf o;
#pragma unroll
  for (int e = 0; e < 8; ++e) o[e] = (bf16_t)src[i + e];
  *(v8bf*)(dst + i) = o;
}

// ---------------------------------------------------------------------------
// vT[b][n][k] = bf16(V[b][k][n]) so the output-GEMM B fragment is per-lane
// contiguous.
// ---------------------------------------------------------------------------
__global__ void k_vT(const float* __restrict__ v, bf16_t* __restrict__ vT) {
  long i = (long)blockIdx.x * blockDim.x + threadIdx.x;  // over B*HID*S
  int k = (int)(i % S_);
  int n = (int)((i / S_) % HID_);
  int b = (int)(i / ((long)S_ * HID_));
  vT[i] = (bf16_t)v[((long)b * S_ + k) * HID_ + n];
}

// ---------------------------------------------------------------------------
// Projection GEMM: out = bf16(scale * (x @ W^T + bias)), all-bf16 inputs.
// Wave computes 32(M) x 64(N): 2 A-frags share 4 B-frags -> 8 WMMA / 6 loads.
// Double-buffered K-loop. Block = 8 waves (256 rows); grid (16, 16).
// ---------------------------------------------------------------------------
__global__ __launch_bounds__(256) void k_proj(const bf16_t* __restrict__ x,
                                              const bf16_t* __restrict__ w,
                                              const float* __restrict__ bias,
                                              float scale,
                                              bf16_t* __restrict__ out) {
  const int lane = threadIdx.x & 31;
  const int wave = threadIdx.x >> 5;
  const int m0 = blockIdx.x * 256 + wave * 32;
  const int n0 = blockIdx.y * 64;

  v8f acc[2][4] = {};
  v16bf a0 = a_frag_bf16(x, HID_, m0,      0, lane);
  v16bf a1 = a_frag_bf16(x, HID_, m0 + 16, 0, lane);
  v16bf bm[4];
#pragma unroll
  for (int t = 0; t < 4; ++t)
    bm[t] = b_frag_rowk_bf16(w, HID_, n0 + 16 * t, 0, lane);

  for (int k = 32; k < HID_; k += 32) {
    // issue next step's loads, then consume current fragments
    v16bf a0n = a_frag_bf16(x, HID_, m0,      k, lane);
    v16bf a1n = a_frag_bf16(x, HID_, m0 + 16, k, lane);
    v16bf bn[4];
#pragma unroll
    for (int t = 0; t < 4; ++t)
      bn[t] = b_frag_rowk_bf16(w, HID_, n0 + 16 * t, k, lane);
#pragma unroll
    for (int t = 0; t < 4; ++t) {
      acc[0][t] = wmma_bf16(a0, bm[t], acc[0][t]);
      acc[1][t] = wmma_bf16(a1, bm[t], acc[1][t]);
    }
    a0 = a0n; a1 = a1n;
#pragma unroll
    for (int t = 0; t < 4; ++t) bm[t] = bn[t];
  }
#pragma unroll
  for (int t = 0; t < 4; ++t) {               // epilogue step
    acc[0][t] = wmma_bf16(a0, bm[t], acc[0][t]);
    acc[1][t] = wmma_bf16(a1, bm[t], acc[1][t]);
  }

  const int n = lane & 15, hi = lane >> 4;
#pragma unroll
  for (int g = 0; g < 2; ++g)
#pragma unroll
    for (int t = 0; t < 4; ++t) {
      const int col = n0 + 16 * t + n;
      const float bc = bias[col];
#pragma unroll
      for (int j = 0; j < 8; ++j) {
        const int m = m0 + 16 * g + j + 8 * hi;
        out[(long)m * HID_ + col] = (bf16_t)((acc[g][t][j] + bc) * scale);
      }
    }
}

// ---------------------------------------------------------------------------
// Fused scores + softmax + head-sum.
// One block per (b, 16 query rows); 8 waves each own 8 of the 64 16x16 score
// tiles per head, double-buffering the K-tile fragment loads. LDS columns are
// XOR-swizzled so both the WMMA scatter (row pairs {j, j+8}) and the softmax
// reads (row pairs {r, r+1}) are bank-conflict-free.
// psum = bf16(sum_h probs / NH).
// ---------------------------------------------------------------------------
__device__ inline int swz(int row, int col) {
  return col ^ ((row & 1) << 4) ^ ((row & 8) << 2);
}

__global__ __launch_bounds__(256) void k_attn(const bf16_t* __restrict__ qp,
                                              const bf16_t* __restrict__ kp,
                                              const float* __restrict__ mask,
                                              bf16_t* __restrict__ psum) {
  __shared__ float sc[16][S_];                   // 64 KB score tile
  const int b   = blockIdx.y;
  const int q0  = blockIdx.x * 16;
  const int tid = threadIdx.x;
  const int lane = tid & 31;
  const int wave = tid >> 5;
  const int srow = tid >> 4;                     // softmax role: row 0..15
  const int sl   = tid & 15;                     // column interleave slot

  float acc[64];
#pragma unroll
  for (int j = 0; j < 64; ++j) acc[j] = 0.f;

  const int rowQ = b * S_ + q0;
  const int ktb  = wave * 8;                     // this wave's 8 key tiles
  for (int h = 0; h < NH_; ++h) {
    const int d0 = h * HD_;
    const v16bf aq0 = a_frag_bf16(qp, HID_, rowQ, d0, lane);
    const v16bf aq1 = a_frag_bf16(qp, HID_, rowQ, d0 + 32, lane);
    v16bf b0 = b_frag_rowk_bf16(kp, HID_, b * S_ + ktb * 16, d0, lane);
    v16bf b1 = b_frag_rowk_bf16(kp, HID_, b * S_ + ktb * 16, d0 + 32, lane);
#pragma unroll
    for (int i = 0; i < 8; ++i) {
      v16bf b0n = b0, b1n = b1;
      if (i < 7) {
        const int rowK = b * S_ + (ktb + i + 1) * 16;
        b0n = b_frag_rowk_bf16(kp, HID_, rowK, d0, lane);
        b1n = b_frag_rowk_bf16(kp, HID_, rowK, d0 + 32, lane);
      }
      v8f c = {};
      c = wmma_bf16(aq0, b0, c);
      c = wmma_bf16(aq1, b1, c);
      const int n = lane & 15, hi = lane >> 4;
#pragma unroll
      for (int j = 0; j < 8; ++j) {
        const int m = j + 8 * hi;
        sc[m][swz(m, (ktb + i) * 16 + n)] = c[j];
      }
      b0 = b0n; b1 = b1n;
    }
    __syncthreads();

    // softmax over row `srow`; this thread covers cols sl + 16*j, j=0..63
    float sv[64];
    float mloc = -3.402823466e+38f;
#pragma unroll
    for (int j = 0; j < 64; ++j) {
      const int col = sl + 16 * j;
      const float s = sc[srow][swz(srow, col)] + mask[b * S_ + col];
      sv[j] = s;
      mloc = fmaxf(mloc, s);
    }
#pragma unroll
    for (int off = 8; off >= 1; off >>= 1)
      mloc = fmaxf(mloc, __shfl_xor(mloc, off, 16));
    float ssum = 0.f;
#pragma unroll
    for (int j = 0; j < 64; ++j) {
      const float e = __expf(sv[j] - mloc);
      sv[j] = e;
      ssum += e;
    }
#pragma unroll
    for (int off = 8; off >= 1; off >>= 1)
      ssum += __shfl_xor(ssum, off, 16);
    const float inv = 1.f / ssum;
#pragma unroll
    for (int j = 0; j < 64; ++j) acc[j] += sv[j] * inv;
    __syncthreads();                             // sc reused next head
  }

  const float invNH = 1.0f / (float)NH_;
  bf16_t* dst = psum + ((long)b * S_ + (q0 + srow)) * S_;
#pragma unroll
  for (int j = 0; j < 64; ++j) dst[sl + 16 * j] = (bf16_t)(acc[j] * invNH);
}

// ---------------------------------------------------------------------------
// Output GEMM: out[b] = psum[b] @ V[b]  (f32 straight to d_out).
// Wave computes 32(M) x 64(N), double-buffered K-loop; grid (4, 16, 4).
// ---------------------------------------------------------------------------
__global__ __launch_bounds__(256) void k_out(const bf16_t* __restrict__ psum,
                                             const bf16_t* __restrict__ vT,
                                             float* __restrict__ out) {
  const int lane = threadIdx.x & 31;
  const int wave = threadIdx.x >> 5;
  const int b  = blockIdx.z;
  const int m0 = blockIdx.x * 256 + wave * 32;   // query row
  const int n0 = blockIdx.y * 64;                // HID column
  const bf16_t* A  = psum + (long)b * S_ * S_;
  const bf16_t* Bv = vT   + (long)b * HID_ * S_;

  v8f acc[2][4] = {};
  v16bf a0 = a_frag_bf16(A, S_, m0,      0, lane);
  v16bf a1 = a_frag_bf16(A, S_, m0 + 16, 0, lane);
  v16bf bm[4];
#pragma unroll
  for (int t = 0; t < 4; ++t)
    bm[t] = b_frag_rowk_bf16(Bv, S_, n0 + 16 * t, 0, lane);

  for (int k = 32; k < S_; k += 32) {
    v16bf a0n = a_frag_bf16(A, S_, m0,      k, lane);
    v16bf a1n = a_frag_bf16(A, S_, m0 + 16, k, lane);
    v16bf bn[4];
#pragma unroll
    for (int t = 0; t < 4; ++t)
      bn[t] = b_frag_rowk_bf16(Bv, S_, n0 + 16 * t, k, lane);
#pragma unroll
    for (int t = 0; t < 4; ++t) {
      acc[0][t] = wmma_bf16(a0, bm[t], acc[0][t]);
      acc[1][t] = wmma_bf16(a1, bm[t], acc[1][t]);
    }
    a0 = a0n; a1 = a1n;
#pragma unroll
    for (int t = 0; t < 4; ++t) bm[t] = bn[t];
  }
#pragma unroll
  for (int t = 0; t < 4; ++t) {                  // epilogue step
    acc[0][t] = wmma_bf16(a0, bm[t], acc[0][t]);
    acc[1][t] = wmma_bf16(a1, bm[t], acc[1][t]);
  }

  const int n = lane & 15, hi = lane >> 4;
#pragma unroll
  for (int g = 0; g < 2; ++g)
#pragma unroll
    for (int t = 0; t < 4; ++t) {
      const int col = n0 + 16 * t + n;
#pragma unroll
      for (int j = 0; j < 8; ++j) {
        const int m = m0 + 16 * g + j + 8 * hi;
        out[((long)b * S_ + m) * HID_ + col] = acc[g][t][j];
      }
    }
}

// ---------------------------------------------------------------------------
extern "C" void kernel_launch(void* const* d_in, const int* in_sizes, int n_in,
                              void* d_out, int out_size, void* d_ws, size_t ws_size,
                              hipStream_t stream) {
  (void)in_sizes; (void)n_in; (void)out_size; (void)ws_size;
  const float* mask  = (const float*)d_in[0];  // [B,1,1,S]
  const float* query = (const float*)d_in[1];  // [B,S,HID]
  const float* key   = (const float*)d_in[2];
  const float* value = (const float*)d_in[3];
  const float* Wq    = (const float*)d_in[4];  // [HID,HID]
  const float* bq    = (const float*)d_in[5];
  const float* Wk    = (const float*)d_in[6];
  const float* bk    = (const float*)d_in[7];
  float* out = (float*)d_out;

  char* ws = (char*)d_ws;
  const size_t NTOK = (size_t)B_ * S_ * HID_;          // 4M elements
  const size_t NW   = (size_t)HID_ * HID_;             // 1M elements
  // Layout (bytes): xq[8M] | xk[8M] | wq[2M] | wk[2M] | qp[8M] | kp[8M] | vT[8M]
  // psum aliases xq: xq is dead after the first k_proj; psum is first written
  // by k_attn which runs strictly later on the same stream.
  bf16_t* xq   = (bf16_t*)(ws);
  bf16_t* xk   = (bf16_t*)(ws + NTOK * 2);
  bf16_t* wqb  = (bf16_t*)(ws + NTOK * 4);
  bf16_t* wkb  = (bf16_t*)(ws + NTOK * 4 + NW * 2);
  bf16_t* qp   = (bf16_t*)(ws + NTOK * 4 + NW * 4);
  bf16_t* kp   = (bf16_t*)(ws + NTOK * 6 + NW * 4);
  bf16_t* vT   = (bf16_t*)(ws + NTOK * 8 + NW * 4);
  bf16_t* psum = xq;                                   // alias (see above)

  // 0) one-shot bf16 staging: X, W (straight), V (transposed per batch)
  k_cvt<<<(int)(NTOK / 2048), 256, 0, stream>>>(query, xq);
  k_cvt<<<(int)(NTOK / 2048), 256, 0, stream>>>(key,   xk);
  k_cvt<<<(int)(NW   / 2048), 256, 0, stream>>>(Wq,    wqb);
  k_cvt<<<(int)(NW   / 2048), 256, 0, stream>>>(Wk,    wkb);
  k_vT <<<(int)(NTOK / 256),  256, 0, stream>>>(value, vT);

  // 1) Q/K projections (score scale 1/sqrt(HD)=0.125 folded into Q)
  dim3 gp(16, 16);
  k_proj<<<gp, 256, 0, stream>>>(xq, wqb, bq, 0.125f, qp);
  k_proj<<<gp, 256, 0, stream>>>(xk, wkb, bk, 1.0f,   kp);

  // 2) scores + softmax + head-sum -> psum = (sum_h probs)/NH
  dim3 ga(S_ / 16, B_);
  k_attn<<<ga, 256, 0, stream>>>(qp, kp, mask, psum);

  // 3) out = psum @ V
  dim3 go(4, 16, 4);
  k_out<<<go, 256, 0, stream>>>(psum, vT, out);
}
